// GraphTransformer_net_64295660421250
// MI455X (gfx1250) — matrix operand order repaired and checked by reading, not compiled, not run on hardware
//
#include <hip/hip_runtime.h>

#ifndef USE_TDM
#define USE_TDM 1
#endif

// ---------------- constants ----------------
constexpr int cN   = 20000;   // nodes
constexpr int cE   = 200000;  // edges
constexpr int cH   = 128;     // HID
constexpr int cNH  = 8;       // heads
constexpr int cDH  = 16;      // head dim
constexpr int cL   = 7;       // layers
constexpr int cMIN = 285;     // MLP_IN
constexpr int cMP  = 288;     // padded MLP width (multiple of 16/32)
constexpr float cEPS = 1e-5f;
constexpr int TB = 256;       // threads per block (8 waves, wave32)

typedef __bf16 bf16;
typedef __attribute__((ext_vector_type(16))) __bf16 v16bf;
typedef __attribute__((ext_vector_type(8)))  __bf16 v8bf;
typedef __attribute__((ext_vector_type(8)))  float  v8f;
typedef __attribute__((ext_vector_type(4)))  unsigned int v4u;
typedef __attribute__((ext_vector_type(8)))  int v8i;
typedef __attribute__((ext_vector_type(4)))  int v4i;

// ---------------- helpers ----------------
__device__ __forceinline__ unsigned f2ord(float f) {
  unsigned u = __float_as_uint(f);
  return (u & 0x80000000u) ? ~u : (u | 0x80000000u);
}
__device__ __forceinline__ float ord2f(unsigned u) {
  return (u & 0x80000000u) ? __uint_as_float(u & 0x7fffffffu) : __uint_as_float(~u);
}
__device__ __forceinline__ v16bf ldA(const bf16* p) {
  v8bf lo = *(const v8bf*)p;
  v8bf hi = *(const v8bf*)(p + 16);
  return __builtin_shufflevector(lo, hi, 0, 1, 2, 3, 4, 5, 6, 7,
                                 8, 9, 10, 11, 12, 13, 14, 15);
}

// ---------------- node embedding: h = nf @ W_emb + b_emb ----------------
__global__ void k_embed(const float* __restrict__ nf, const float* __restrict__ We,
                        const float* __restrict__ be, float* __restrict__ h) {
  int tid = blockIdx.x * blockDim.x + threadIdx.x;
  if (tid >= cN * cH) return;
  int n = tid >> 7, c = tid & 127;
  h[tid] = nf[2 * n] * We[c] + nf[2 * n + 1] * We[cH + c] + be[c];
}

// --------- weight transpose+pad to bf16 [Nc][K] (B-fragment friendly) ----------
__global__ void k_wtrans(const float* __restrict__ W, bf16* __restrict__ Bt,
                         int srcK, int srcN, int K, int Nc) {
  int tid = blockIdx.x * blockDim.x + threadIdx.x;
  if (tid >= K * Nc) return;
  int n = tid / K, k = tid - n * K;
  float v = (k < srcK && n < srcN) ? W[k * srcN + n] : 0.f;
  Bt[tid] = (bf16)v;
}

// ---------------- layernorm (one wave per row) -> bf16 ----------------
__global__ void k_ln(const float* __restrict__ h, const float* __restrict__ g,
                     const float* __restrict__ b, bf16* __restrict__ x) {
  int wid  = (blockIdx.x * blockDim.x + threadIdx.x) >> 5;
  int lane = threadIdx.x & 31;
  if (wid >= cN) return;
  const float* hr = h + (size_t)wid * cH;
  float4 v = ((const float4*)hr)[lane];
  float s = v.x + v.y + v.z + v.w;
  #pragma unroll
  for (int m = 16; m >= 1; m >>= 1) s += __shfl_xor(s, m, 32);
  float mu = s * (1.f / cH);
  float dx = v.x - mu, dy = v.y - mu, dz = v.z - mu, dw = v.w - mu;
  float q = dx * dx + dy * dy + dz * dz + dw * dw;
  #pragma unroll
  for (int m = 16; m >= 1; m >>= 1) q += __shfl_xor(q, m, 32);
  float rs = rsqrtf(q * (1.f / cH) + cEPS);
  int c = lane * 4;
  bf16* xr = x + (size_t)wid * cH;
  xr[c + 0] = (bf16)(dx * rs * g[c + 0] + b[c + 0]);
  xr[c + 1] = (bf16)(dy * rs * g[c + 1] + b[c + 1]);
  xr[c + 2] = (bf16)(dz * rs * g[c + 2] + b[c + 2]);
  xr[c + 3] = (bf16)(dw * rs * g[c + 3] + b[c + 3]);
}

// ============ WMMA bf16 GEMM, TDM/LDS-staged B panel, MR=2 x NT tiles/wave =====
// Block = 8 waves. Wave w computes 32-row strip (blockIdx.x*8 + w) x NT column
// tiles starting at column blockIdx.y*NT*16. B panel [NT*16 cols][K] lives in
// LDS (B^T layout, K contiguous); K is a template constant so every DS access
// uses an immediate offset. Each B fragment feeds two WMMAs (two row strips).
// MODE 0: outF = acc + bias
// MODE 1: outF = res + acc + bias   (res may alias outF)
// MODE 2: outB = relu(acc + bias)   (bf16 out)
template <int MODE, int NT, int K>
__global__ void __launch_bounds__(TB, 1)
k_gemm(const bf16* __restrict__ A, const bf16* __restrict__ Bt,
       const float* __restrict__ bias, int biasN,
       const float* __restrict__ res, float* __restrict__ outF,
       bf16* __restrict__ outB, int M, int Nc) {
  constexpr int KS = K / 32;
  constexpr int PANEL = NT * 16 * K;    // bf16 elements
  extern __shared__ bf16 sB[];
  const int lane = threadIdx.x & 31;
  const int wv   = threadIdx.x >> 5;
  const int n0   = blockIdx.y * (NT * 16);
  const bf16* gpanel = Bt + (size_t)n0 * K;

#if USE_TDM
  if (wv == 0) {
    // Tensor Data Mover: 1-row 2D descriptor copying the contiguous panel.
    unsigned ldsaddr = (unsigned)(size_t)(const void*)sB;
    unsigned long long ga = (unsigned long long)(size_t)gpanel;
    v4u g0 = {1u,                                            // count=1 (valid)
              ldsaddr,                                       // lds_addr
              (unsigned)(ga & 0xffffffffu),                  // global_addr[31:0]
              (unsigned)(((ga >> 32) & 0x01ffffffu) | 0x80000000u)}; // addr[56:32] | type=2
    v8i g1 = {(int)0x00010000,                               // data_size=1 (2 bytes)
              (int)((PANEL & 0xffff) << 16),                 // tensor_dim0[15:0]
              (int)(((unsigned)PANEL >> 16) | (1u << 16)),   // tensor_dim0[31:16], tensor_dim1=1
              (int)((PANEL & 0xffff) << 16),                 // tile_dim0 = PANEL
              (int)1,                                        // tile_dim1 = 1
              (int)PANEL,                                    // tensor_dim0_stride[31:0]
              0, 0};
    v4i z4 = {0, 0, 0, 0};
#if __clang_major__ >= 23
    v8i z8 = {0, 0, 0, 0, 0, 0, 0, 0};
    __builtin_amdgcn_tensor_load_to_lds(g0, g1, z4, z4, z8, 0);
#else
    __builtin_amdgcn_tensor_load_to_lds(g0, g1, z4, z4, 0);
#endif
    __builtin_amdgcn_s_wait_tensorcnt(0);
  }
#else
  for (int i = threadIdx.x * 8; i < PANEL; i += TB * 8)
    *(v8bf*)&sB[i] = *(const v8bf*)&gpanel[i];
#endif
  __syncthreads();

  const int rt = blockIdx.x * 8 + wv;   // 32-row strip index
  if (rt >= (M >> 5)) return;           // wave-uniform: EXEC stays all-ones
  const int hl = lane >> 4, nl = lane & 15;

  // A frag (16x32 bf16): lane row; e<8 -> K=8*hl+e ; e>=8 -> K=16+8*hl+(e-8)
  const bf16* arow0 = A + (size_t)((rt << 5) + nl) * K + 8 * hl;
  const bf16* arow1 = arow0 + (size_t)16 * K;
  // B frag base: lane col = nl; e<8 -> K=8*hl+e ; e>=8 -> K=16+8*hl+(e-8)
  const bf16* bbase = sB + nl * K + 8 * hl;

  v8f acc0[NT], acc1[NT];
  const v8f vzero = {0.f, 0.f, 0.f, 0.f, 0.f, 0.f, 0.f, 0.f};
  #pragma unroll
  for (int t = 0; t < NT; ++t) { acc0[t] = vzero; acc1[t] = vzero; }

  #pragma unroll
  for (int ks = 0; ks < KS; ++ks) {
    __builtin_prefetch(arow0 + ks * 32 + 512, 0, 1);   // global_prefetch_b8
    v16bf a0 = ldA(arow0 + ks * 32);
    v16bf a1 = ldA(arow1 + ks * 32);
    #pragma unroll
    for (int t = 0; t < NT; ++t) {
      v16bf bm = ldA(bbase + t * 16 * K + ks * 32);  // imm DS offsets
      acc0[t] = __builtin_amdgcn_wmma_f32_16x16x32_bf16(false, a0, false, bm,
                                                        (short)0, acc0[t], false, false);
      acc1[t] = __builtin_amdgcn_wmma_f32_16x16x32_bf16(false, a1, false, bm,
                                                        (short)0, acc1[t], false, false);
    }
  }

  #pragma unroll
  for (int t = 0; t < NT; ++t) {
    int n = n0 + t * 16 + nl;
    float bv = (n < biasN) ? bias[n] : 0.f;
    #pragma unroll
    for (int r = 0; r < 8; ++r) {
      int gm0 = (rt << 5) + r + 8 * hl;  // C VGPR r -> M = r + 8*half
      int gm1 = gm0 + 16;
      size_t oi0 = (size_t)gm0 * Nc + n;
      size_t oi1 = (size_t)gm1 * Nc + n;
      float v0 = acc0[t][r] + bv;
      float v1 = acc1[t][r] + bv;
      if (MODE == 0)      { outF[oi0] = v0; outF[oi1] = v1; }
      else if (MODE == 1) { outF[oi0] = res[oi0] + v0; outF[oi1] = res[oi1] + v1; }
      else { outB[oi0] = (bf16)fmaxf(v0, 0.f); outB[oi1] = (bf16)fmaxf(v1, 0.f); }
    }
  }
}

// ---------------- attention scratch init ----------------
__global__ void k_attn_init(float* __restrict__ attn, float* __restrict__ den,
                            unsigned* __restrict__ mord) {
  int tid = blockIdx.x * blockDim.x + threadIdx.x;
  if (tid < cN * cH) attn[tid] = 0.f;
  if (tid < cN * cNH) { den[tid] = 0.f; mord[tid] = 0u; }
}

// ---------------- edge scores + segment max ----------------
__global__ void k_scores(const float* __restrict__ Q, const float* __restrict__ Km,
                         const int* __restrict__ src, const int* __restrict__ dst,
                         float* __restrict__ sc, unsigned* __restrict__ mord) {
  int tid = blockIdx.x * blockDim.x + threadIdx.x;
  if (tid >= cE * cNH) return;
  int e = tid >> 3, hd = tid & 7;
  int d = dst[e], s = src[e];
  const float4* q = (const float4*)(Q + (size_t)d * cH + hd * cDH);
  const float4* k = (const float4*)(Km + (size_t)s * cH + hd * cDH);
  float a = 0.f;
  #pragma unroll
  for (int i = 0; i < 4; ++i) {
    float4 qi = q[i], ki = k[i];
    a += qi.x * ki.x + qi.y * ki.y + qi.z * ki.z + qi.w * ki.w;
  }
  a *= 0.25f;                               // 1/sqrt(DH), DH=16
  sc[tid] = a;
  atomicMax(mord + (size_t)d * cNH + hd, f2ord(a));
}

// ---------------- exp(score - max) + segment sum ----------------
__global__ void k_expsum(const int* __restrict__ dst, const unsigned* __restrict__ mord,
                         float* __restrict__ sc, float* __restrict__ den) {
  int tid = blockIdx.x * blockDim.x + threadIdx.x;
  if (tid >= cE * cNH) return;
  int e = tid >> 3, hd = tid & 7;
  float mm = ord2f(mord[(size_t)dst[e] * cNH + hd]);
  float ex = __expf(sc[tid] - mm);
  sc[tid] = ex;
  atomicAdd(den + (size_t)dst[e] * cNH + hd, ex);
}

// ---------------- weighted V scatter ----------------
__global__ void k_scatter(const int* __restrict__ src, const int* __restrict__ dst,
                          const float* __restrict__ sc, const float* __restrict__ den,
                          const float* __restrict__ V, float* __restrict__ attn) {
  int tid = blockIdx.x * blockDim.x + threadIdx.x;
  if (tid >= cE * cNH) return;
  int e = tid >> 3, hd = tid & 7;
  int d = dst[e];
  float w = sc[tid] / den[(size_t)d * cNH + hd];
  const float* v = V + (size_t)src[e] * cH + hd * cDH;
  float* a = attn + (size_t)d * cH + hd * cDH;
  #pragma unroll
  for (int i = 0; i < cDH; ++i) atomicAdd(a + i, w * v[i]);
}

// ---------------- f32 -> bf16 ----------------
__global__ void k_f2bf(const float* __restrict__ in, bf16* __restrict__ out, int n) {
  int tid = blockIdx.x * blockDim.x + threadIdx.x;
  if (tid < n) out[tid] = (bf16)in[tid];
}

// ---------------- edge classifier input y0 [E, 288] (padded) ----------------
__global__ void k_build_y(const float* __restrict__ h, const float* __restrict__ curr,
                          const float* __restrict__ pay, const float* __restrict__ nums,
                          const int* __restrict__ src, const int* __restrict__ dst,
                          const int* __restrict__ pc, const int* __restrict__ rc,
                          const int* __restrict__ pf, bf16* __restrict__ y) {
  int tid = blockIdx.x * blockDim.x + threadIdx.x;
  if (tid >= cE * cMP) return;
  int e = tid / cMP, j = tid - e * cMP;
  float v;
  if (j < 128)      v = h[(size_t)src[e] * cH + j];
  else if (j < 256) v = h[(size_t)dst[e] * cH + (j - 128)];
  else if (j < 264) v = curr[pc[e] * 8 + (j - 256)];
  else if (j < 272) v = curr[rc[e] * 8 + (j - 264)];
  else if (j < 280) v = pay[pf[e] * 8 + (j - 272)];
  else if (j < 285) v = nums[(size_t)e * 5 + (j - 280)];
  else              v = 0.f;
  y[tid] = (bf16)v;
}

// ---------------- output layer [E,285] @ [285,2] ----------------
__global__ void k_final(const bf16* __restrict__ y, const float* __restrict__ Wo,
                        const float* __restrict__ bo, float* __restrict__ out) {
  int e = blockIdx.x * blockDim.x + threadIdx.x;
  if (e >= cE) return;
  const bf16* yr = y + (size_t)e * cMP;
  float s0 = bo[0], s1 = bo[1];
  for (int j = 0; j < cMIN; ++j) {
    float yv = (float)yr[j];
    s0 += yv * Wo[2 * j];
    s1 += yv * Wo[2 * j + 1];
  }
  out[2 * e] = s0;
  out[2 * e + 1] = s1;
}

// ---------------- host orchestration ----------------
static inline size_t alignup(size_t x) { return (x + 255) & ~(size_t)255; }
static inline int nblk(long long n) { return (int)((n + TB - 1) / TB); }

extern "C" void kernel_launch(void* const* d_in, const int* in_sizes, int n_in,
                              void* d_out, int out_size, void* d_ws, size_t ws_size,
                              hipStream_t stream) {
  (void)in_sizes; (void)n_in; (void)out_size; (void)ws_size;
  const float* node_feats = (const float*)d_in[0];
  const float* numericals = (const float*)d_in[1];
  const float* W_emb = (const float*)d_in[2];
  const float* b_emb = (const float*)d_in[3];
  const float* WQ = (const float*)d_in[4];  const float* bQ = (const float*)d_in[5];
  const float* WK = (const float*)d_in[6];  const float* bK = (const float*)d_in[7];
  const float* WV = (const float*)d_in[8];  const float* bV = (const float*)d_in[9];
  const float* WO = (const float*)d_in[10]; const float* bO = (const float*)d_in[11];
  const float* ln1_g = (const float*)d_in[12]; const float* ln1_b = (const float*)d_in[13];
  const float* ln2_g = (const float*)d_in[14]; const float* ln2_b = (const float*)d_in[15];
  const float* W1 = (const float*)d_in[16]; const float* b1 = (const float*)d_in[17];
  const float* W2 = (const float*)d_in[18]; const float* b2 = (const float*)d_in[19];
  const float* curr_emb = (const float*)d_in[20]; const float* pay_emb = (const float*)d_in[21];
  const float* mlp_Wh = (const float*)d_in[22]; const float* mlp_bh = (const float*)d_in[23];
  const float* mlp_Wo = (const float*)d_in[24]; const float* mlp_bo = (const float*)d_in[25];
  const int* src = (const int*)d_in[26]; const int* dst = (const int*)d_in[27];
  const int* pc  = (const int*)d_in[28]; const int* rc  = (const int*)d_in[29];
  const int* pf  = (const int*)d_in[30];

  char* ws = (char*)d_ws;
  size_t off = 0;
  auto take = [&](size_t bytes) -> void* {
    void* p = ws + off;
    off = alignup(off + bytes);
    return p;
  };
  float*    h     = (float*)take((size_t)cN * cH * 4);
  bf16*     xb    = (bf16*) take((size_t)cN * cH * 2);
  float*    Qb    = (float*)take((size_t)cN * cH * 4);
  float*    Kb    = (float*)take((size_t)cN * cH * 4);
  float*    Vb    = (float*)take((size_t)cN * cH * 4);
  float*    attn  = (float*)take((size_t)cN * cH * 4);
  bf16*     attnb = (bf16*) take((size_t)cN * cH * 2);
  bf16*     tb    = (bf16*) take((size_t)cN * cH * 2);
  float*    sc    = (float*)take((size_t)cE * cNH * 4);
  unsigned* mord  = (unsigned*)take((size_t)cN * cNH * 4);
  float*    den   = (float*)take((size_t)cN * cNH * 4);
  bf16*     wl    = (bf16*) take((size_t)cL * 6 * cH * cH * 2);  // transposed layer weights
  bf16*     wm    = (bf16*) take((size_t)3 * cMP * cMP * 2);     // transposed+padded MLP weights
  bf16*     ya    = (bf16*) take((size_t)cE * cMP * 2);
  bf16*     yb    = (bf16*) take((size_t)cE * cMP * 2);

  // --- node embedding ---
  k_embed<<<nblk((long long)cN * cH), TB, 0, stream>>>(node_feats, W_emb, b_emb, h);

  // --- weight prep (transpose to B-fragment layout, bf16) ---
  const int wsz = cH * cH;
  for (int l = 0; l < cL; ++l) {
    const float* srcs[6] = {WQ + (size_t)l * wsz, WK + (size_t)l * wsz, WV + (size_t)l * wsz,
                            WO + (size_t)l * wsz, W1 + (size_t)l * wsz, W2 + (size_t)l * wsz};
    for (int w = 0; w < 6; ++w)
      k_wtrans<<<nblk(wsz), TB, 0, stream>>>(srcs[w], wl + (size_t)(l * 6 + w) * wsz,
                                             cH, cH, cH, cH);
  }
  for (int i = 0; i < 3; ++i)
    k_wtrans<<<nblk(cMP * cMP), TB, 0, stream>>>(mlp_Wh + (size_t)i * cMIN * cMIN,
                                                 wm + (size_t)i * cMP * cMP,
                                                 cMIN, cMIN, cMP, cMP);

  // Layer GEMMs: NT=4 (64-col groups, gridDim.y=2), 32-row strips,
  // panel = 64x128 bf16 = 16 KB LDS. acc = 64 VGPRs/wave -> no spills.
  const int gblk = (cN / 32 + 7) / 8;            // 625 strips -> 79 blocks
  const dim3 ggrid(gblk, 2);
  const size_t gshm = (size_t)4 * 16 * cH * 2;   // 16384 B

  // --- transformer layers ---
  for (int l = 0; l < cL; ++l) {
    const bf16* WQt = wl + (size_t)(l * 6 + 0) * wsz;
    const bf16* WKt = wl + (size_t)(l * 6 + 1) * wsz;
    const bf16* WVt = wl + (size_t)(l * 6 + 2) * wsz;
    const bf16* WOt = wl + (size_t)(l * 6 + 3) * wsz;
    const bf16* W1t = wl + (size_t)(l * 6 + 4) * wsz;
    const bf16* W2t = wl + (size_t)(l * 6 + 5) * wsz;

    k_ln<<<nblk((long long)cN * 32), TB, 0, stream>>>(h, ln1_g + l * cH, ln1_b + l * cH, xb);
    k_gemm<0, 4, cH><<<ggrid, TB, gshm, stream>>>(xb, WQt, bQ + l * cH, cH, nullptr, Qb, nullptr, cN, cH);
    k_gemm<0, 4, cH><<<ggrid, TB, gshm, stream>>>(xb, WKt, bK + l * cH, cH, nullptr, Kb, nullptr, cN, cH);
    k_gemm<0, 4, cH><<<ggrid, TB, gshm, stream>>>(xb, WVt, bV + l * cH, cH, nullptr, Vb, nullptr, cN, cH);

    k_attn_init<<<nblk((long long)cN * cH), TB, 0, stream>>>(attn, den, mord);
    k_scores<<<nblk((long long)cE * cNH), TB, 0, stream>>>(Qb, Kb, src, dst, sc, mord);
    k_expsum<<<nblk((long long)cE * cNH), TB, 0, stream>>>(dst, mord, sc, den);
    k_scatter<<<nblk((long long)cE * cNH), TB, 0, stream>>>(src, dst, sc, den, Vb, attn);
    k_f2bf<<<nblk((long long)cN * cH), TB, 0, stream>>>(attn, attnb, cN * cH);

    k_gemm<1, 4, cH><<<ggrid, TB, gshm, stream>>>(attnb, WOt, bO + l * cH, cH, h, h, nullptr, cN, cH);
    k_ln<<<nblk((long long)cN * 32), TB, 0, stream>>>(h, ln2_g + l * cH, ln2_b + l * cH, xb);
    k_gemm<2, 4, cH><<<ggrid, TB, gshm, stream>>>(xb, W1t, b1 + l * cH, cH, nullptr, nullptr, tb, cN, cH);
    k_gemm<1, 4, cH><<<ggrid, TB, gshm, stream>>>(tb, W2t, b2 + l * cH, cH, h, h, nullptr, cN, cH);
  }

  // --- edge classifier ---
  k_build_y<<<nblk((long long)cE * cMP), TB, 0, stream>>>(h, curr_emb, pay_emb, numericals,
                                                          src, dst, pc, rc, pf, ya);
  // MLP GEMMs: NT=6 (96-col groups, gridDim.y=3), panel = 96x288 bf16 = 54 KB
  const int mblk = (cE / 32 + 7) / 8;            // 6250 strips -> 782 blocks
  const dim3 mgrid(mblk, 3);
  const size_t mshm = (size_t)6 * 16 * cMP * 2;  // 55296 B
  bf16* yin = ya;
  bf16* yout = yb;
  for (int i = 0; i < 3; ++i) {
    k_gemm<2, 6, cMP><<<mgrid, TB, mshm, stream>>>(yin, wm + (size_t)i * cMP * cMP, mlp_bh + i * cMIN,
                                                   cMIN, nullptr, nullptr, yout, cE, cMP);
    bf16* t = yin; yin = yout; yout = t;
  }
  k_final<<<nblk(cE), TB, 0, stream>>>(yin, mlp_Wo, mlp_bo, (float*)d_out);
}
